// GNN_Saturation_Pressure_30451318129171
// MI455X (gfx1250) — matrix operand order repaired and checked by reading, not compiled, not run on hardware
//
#include <hip/hip_runtime.h>

#define N_NODES  50000
#define N_EDGES  800000
#define N_GRAPHS 1024
#define M_PAD    50048   // 391 * 128  (node rows padded to GEMM tile multiple)
#define KP_MAX   448     // max padded K across all layers

typedef __attribute__((ext_vector_type(16))) __bf16 v16bf;
typedef __attribute__((ext_vector_type(8)))  float  v8f;

typedef union { uint4 q[2]; v16bf v; } Frag;

__device__ __forceinline__ unsigned short f2bf(float f) {
  unsigned int u = __float_as_uint(f);
  u += 0x7FFFu + ((u >> 16) & 1u);          // round-to-nearest-even
  return (unsigned short)(u >> 16);
}

// ---------------------------------------------------------------- utilities
__global__ void k_zero_f32(float* p, int n) {
  int i = blockIdx.x * blockDim.x + threadIdx.x;
  if (i < n) p[i] = 0.0f;
}

__global__ void k_deg_count(const int* __restrict__ dst, float* __restrict__ deg) {
  int e = blockIdx.x * blockDim.x + threadIdx.x;
  if (e < N_EDGES) atomicAdd(&deg[dst[e]], 1.0f);
}

__global__ void k_dinv(const float* __restrict__ deg, float* __restrict__ dinv) {
  int i = blockIdx.x * blockDim.x + threadIdx.x;
  if (i < N_NODES) dinv[i] = rsqrtf(deg[i] + 1.0f);
}

__global__ void k_edge_norm(const int* __restrict__ src, const int* __restrict__ dst,
                            const float* __restrict__ dinv, float* __restrict__ enorm) {
  int e = blockIdx.x * blockDim.x + threadIdx.x;
  if (e < N_EDGES) enorm[e] = dinv[src[e]] * dinv[dst[e]];
}

// fp32 activations -> zero-padded bf16 A (Mpad x Kpad), rows>=M and cols>=Kreal are 0
__global__ void k_conv_a(const float* __restrict__ src, unsigned short* __restrict__ dst,
                         int M, int Kreal, int srcStride, int Kpad, int total) {
  int i = blockIdx.x * blockDim.x + threadIdx.x;
  if (i >= total) return;
  int n = i / Kpad, k = i - n * Kpad;
  float v = (n < M && k < Kreal) ? src[(size_t)n * srcStride + k] : 0.0f;
  dst[i] = f2bf(v);
}

// fp32 weight (Kreal x Nreal row-major) -> transposed zero-padded bf16 Bt (Npad x Kpad)
__global__ void k_conv_wt(const float* __restrict__ w, unsigned short* __restrict__ dst,
                          int Kreal, int Nreal, int Kpad, int Npad) {
  int i = blockIdx.x * blockDim.x + threadIdx.x;
  if (i >= Npad * Kpad) return;
  int n = i / Kpad, k = i - n * Kpad;
  float v = (n < Nreal && k < Kreal) ? w[(size_t)k * Nreal + n] : 0.0f;
  dst[i] = f2bf(v);
}

// ---------------------------------------------------------------- WMMA GEMM
// C[Mpad x Npad] = A[Mpad x Kpad] * Bt^T
// Block (8 waves) -> 128x64 tile. B tile (64 x Kpad bf16, <=56KB) staged in LDS
// by the Tensor Data Mover (TENSOR_LOAD_TO_LDS, TENSORcnt) when available;
// A stream double-buffered from global so WMMAs overlap the load latency.
__global__ __launch_bounds__(256) void k_wmma_gemm(
    const __bf16* __restrict__ A,   // Mpad x Kpad, row-major bf16
    const __bf16* __restrict__ Bt,  // Npad x Kpad, row-major bf16 (weights transposed)
    float* __restrict__ C,          // Mpad x Npad, row-major f32
    int Kpad, int Npad)
{
  __shared__ __align__(16) unsigned short sB[64 * KP_MAX];   // 56 KB of 320 KB WGP LDS

  const int tid   = threadIdx.x;
  const int lane  = tid & 31;
  const int wave  = tid >> 5;
  const int row16 = lane & 15;
  const int hi8   = (lane >> 4) << 3;   // A frag: 0 or 8
  const int boff  = (lane >> 4) << 3;   // B frag: 0 or 8
  const int colb  = blockIdx.y * 64;

#if __has_builtin(__builtin_amdgcn_tensor_load_to_lds) && __has_builtin(__builtin_amdgcn_s_wait_tensorcnt)
  // ---- TDM path: one wave programs the DMA; descriptor per CDNA5 ISA ch.8 ----
  // Keep one visible store to sB so IR never sees a store-free LDS array
  // (the DMA immediately overwrites it).
  if (tid == 0) *(volatile unsigned short*)sB = 0;
  if (wave == 0) {
    typedef unsigned int v4u __attribute__((ext_vector_type(4)));
    typedef int          v8s __attribute__((ext_vector_type(8)));
    typedef int          v4s __attribute__((ext_vector_type(4)));
    unsigned long long ga =
        (unsigned long long)(const void*)(Bt + (size_t)colb * Kpad);  // tile start (bytes)
    unsigned int kp = (unsigned int)Kpad;
    // D# group 0: count=1 | lds_addr(bytes, sB at offset 0) | global_addr | type=2
    v4u g0 = { 1u,                              // [1:0] count=1, valid user descriptor
               0u,                              // [63:32] lds_addr
               (unsigned int)ga,                // [95:64] global_addr lo
               (unsigned int)(ga >> 32) | (2u << 30) };  // [120:96] addr hi | [127:126] type=2
    // D# group 1: data_size=2B; tensor 64 rows x Kpad cols; tile = whole tensor
    v8s g1 = { (int)(1u << 16),                 // [17:16] data_size=1 -> 2 bytes
               (int)((kp & 0xFFFFu) << 16),     // [79:48] tensor_dim0 = Kpad (lo 16)
               (int)((kp >> 16) | (64u << 16)), // tensor_dim0 hi | [111:80] tensor_dim1=64 (lo)
               (int)(kp << 16),                 // tensor_dim1 hi=0 | [127:112] tile_dim0=Kpad
               (int)64,                         // [143:128] tile_dim1=64, [159:144] tile_dim2=0
               (int)kp,                         // [207:160] tensor_dim0_stride = Kpad (lo 32)
               0, 0 };                          // stride hi, tensor_dim1_stride unused (2D)
    v4s gz4 = { 0, 0, 0, 0 };                   // groups 2/3 unused for 2D tensors
    v8s gz8 = { 0, 0, 0, 0, 0, 0, 0, 0 };
    __builtin_amdgcn_tensor_load_to_lds(g0, g1, gz4, gz4, gz8, 0);
    __builtin_amdgcn_s_wait_tensorcnt(0);
  }
  // The TDM wrote LDS behind the compiler's back: escape sB into an asm with a
  // memory clobber (input-only, so reads below keep their LDS address space).
  asm volatile("" : : "v"(&sB[0]) : "memory");
#else
  // ---- fallback: cooperative VGPR round-trip copy ----
  {
    const uint4* gB = reinterpret_cast<const uint4*>(Bt + (size_t)colb * Kpad);
    uint4*       lB = reinterpret_cast<uint4*>(sB);
    int n_u4 = (64 * Kpad) >> 3;                     // 8 bf16 per uint4
    for (int i = tid; i < n_u4; i += 256) lB[i] = gB[i];
  }
#endif
  __syncthreads();

  const size_t arow = (size_t)blockIdx.x * 128 + wave * 16 + row16;
  const __bf16* Ap  = A + arow * Kpad;

  v8f acc[4];
  for (int t = 0; t < 4; ++t)
    for (int r = 0; r < 8; ++r) acc[t][r] = 0.0f;

  // software-pipelined A fragments: issue next k-step loads before current WMMAs
  Frag a_cur, a_nxt;
  a_cur.q[0] = *reinterpret_cast<const uint4*>(Ap + hi8);
  a_cur.q[1] = *reinterpret_cast<const uint4*>(Ap + 16 + hi8);
  a_nxt = a_cur;

  for (int k0 = 0; k0 < Kpad; k0 += 32) {
    int kn = k0 + 32;
    if (kn < Kpad) {
      a_nxt.q[0] = *reinterpret_cast<const uint4*>(Ap + kn + hi8);
      a_nxt.q[1] = *reinterpret_cast<const uint4*>(Ap + kn + 16 + hi8);
    }
#pragma unroll
    for (int t = 0; t < 4; ++t) {
      Frag b;
      const unsigned short* Bp = sB + (unsigned)(t * 16 + row16) * Kpad + k0 + boff;
      b.q[0] = *reinterpret_cast<const uint4*>(Bp);        // ds_load_b128
      b.q[1] = *reinterpret_cast<const uint4*>(Bp + 16);
      acc[t] = __builtin_amdgcn_wmma_f32_16x16x32_bf16(
          false, a_cur.v, false, b.v, (short)0, acc[t], false, false);
    }
    a_cur = a_nxt;
  }

  const size_t orow = (size_t)blockIdx.x * 128 + wave * 16 + hi8;   // D: VGPR r -> M=r(+8)
#pragma unroll
  for (int t = 0; t < 4; ++t) {
    int ocol = colb + t * 16 + row16;
#pragma unroll
    for (int r = 0; r < 8; ++r)
      C[(orow + r) * Npad + ocol] = acc[t][r];
  }
}

// ---------------------------------------------------------------- graph ops
__global__ void k_edge_agg(const float* __restrict__ h, float* __restrict__ agg,
                           const int* __restrict__ src, const int* __restrict__ dst,
                           const float* __restrict__ enorm, int Npad) {
  int i = blockIdx.x * blockDim.x + threadIdx.x;
  int chunks = Npad >> 2;
  if (i >= N_EDGES * chunks) return;
  int e = i / chunks;
  int c = (i - e * chunks) << 2;
  float nrm = enorm[e];
  const float4 hv = *reinterpret_cast<const float4*>(h + (size_t)src[e] * Npad + c);
  float* ap = agg + (size_t)dst[e] * Npad + c;
  atomicAdd(ap + 0, hv.x * nrm);
  atomicAdd(ap + 1, hv.y * nrm);
  atomicAdd(ap + 2, hv.z * nrm);
  atomicAdd(ap + 3, hv.w * nrm);
}

__global__ void k_combine(const float* __restrict__ h, const float* __restrict__ agg,
                          const float* __restrict__ dinv, const float* __restrict__ bias,
                          float* __restrict__ xact, int Npad, int Nreal) {
  int i = blockIdx.x * blockDim.x + threadIdx.x;
  if (i >= N_NODES * Npad) return;
  int n = i / Npad, d = i - n * Npad;
  float v = 0.0f;
  if (d < Nreal) {
    float sn = dinv[n] * dinv[n];
    v = fmaxf(agg[i] + h[i] * sn + bias[d], 0.0f);
  }
  xact[i] = v;
}

__global__ void k_pool(const float* __restrict__ xact, const int* __restrict__ batch,
                       float* __restrict__ g, int Npad) {
  int i = blockIdx.x * blockDim.x + threadIdx.x;
  if (i >= N_NODES * Npad) return;
  int n = i / Npad, d = i - n * Npad;
  atomicAdd(&g[(size_t)batch[n] * Npad + d], xact[i]);
}

__global__ void k_relu(float* p, int n) {
  int i = blockIdx.x * blockDim.x + threadIdx.x;
  if (i < n) p[i] = fmaxf(p[i], 0.0f);
}

__global__ void k_bias_relu(const float* __restrict__ c, const float* __restrict__ bias,
                            float* __restrict__ out, int Npad, int Nreal) {
  int i = blockIdx.x * blockDim.x + threadIdx.x;
  if (i >= N_GRAPHS * Npad) return;
  int d = i % Npad;
  out[i] = (d < Nreal) ? fmaxf(c[i] + bias[d], 0.0f) : 0.0f;
}

__global__ void k_antoine(const float* __restrict__ g, const float* __restrict__ w,
                          const float* __restrict__ b, const float* __restrict__ temp,
                          const float* __restrict__ mean, const float* __restrict__ stdv,
                          float* __restrict__ out) {
  int i = blockIdx.x * blockDim.x + threadIdx.x;
  if (i >= N_GRAPHS) return;
  const float* gv = g + (size_t)i * 128;   // stride 128, 100 real features
  float A = b[0], Bc = b[1], Cc = b[2];
  for (int j = 0; j < 100; ++j) {
    float x = gv[j];
    A  = fmaf(x, w[j * 3 + 0], A);
    Bc = fmaf(x, w[j * 3 + 1], Bc);
    Cc = fmaf(x, w[j * 3 + 2], Cc);
  }
  float logP = A - Bc / (temp[i] + Cc);
  out[i] = (logP - mean[0]) / stdv[0];
}

// ---------------------------------------------------------------- launch
extern "C" void kernel_launch(void* const* d_in, const int* in_sizes, int n_in,
                              void* d_out, int out_size, void* d_ws, size_t ws_size,
                              hipStream_t stream) {
  (void)in_sizes; (void)n_in; (void)out_size; (void)ws_size;
  const float* x     = (const float*)d_in[0];
  const int*   esrc  = (const int*)d_in[1];
  const int*   edst  = esrc + N_EDGES;
  const int*   batch = (const int*)d_in[2];
  const float* temp  = (const float*)d_in[3];
  const float* mean  = (const float*)d_in[4];
  const float* stdv  = (const float*)d_in[5];
  const float* wg[4] = {(const float*)d_in[6],  (const float*)d_in[7],
                        (const float*)d_in[8],  (const float*)d_in[9]};
  const float* bg[4] = {(const float*)d_in[10], (const float*)d_in[11],
                        (const float*)d_in[12], (const float*)d_in[13]};
  const float* wd[4] = {(const float*)d_in[14], (const float*)d_in[15],
                        (const float*)d_in[16], (const float*)d_in[17]};
  const float* bd[4] = {(const float*)d_in[18], (const float*)d_in[19],
                        (const float*)d_in[20], (const float*)d_in[21]};
  const float* want  = (const float*)d_in[22];
  const float* bant  = (const float*)d_in[23];
  float* out = (float*)d_out;

  char* base = (char*)d_ws;
  size_t off = 0;
  auto alloc = [&](size_t bytes) -> char* {
    char* p = base + off;
    off += (bytes + 255) & ~(size_t)255;
    return p;
  };
  float*          deg   = (float*)alloc((size_t)N_NODES * 4);
  float*          dinv  = (float*)alloc((size_t)N_NODES * 4);
  float*          enorm = (float*)alloc((size_t)N_EDGES * 4);
  unsigned short* abf   = (unsigned short*)alloc((size_t)M_PAD * KP_MAX * 2);
  unsigned short* wbf   = (unsigned short*)alloc((size_t)KP_MAX * KP_MAX * 2);
  float*          h     = (float*)alloc((size_t)M_PAD * KP_MAX * 4);
  float*          agg   = (float*)alloc((size_t)M_PAD * KP_MAX * 4);
  float*          xact  = (float*)alloc((size_t)M_PAD * KP_MAX * 4);
  float*          gA    = (float*)alloc((size_t)N_GRAPHS * 320 * 4);
  float*          gB    = (float*)alloc((size_t)N_GRAPHS * 320 * 4);

  const int B = 256;
  auto blocks = [](long long n) { return (unsigned)((n + 255) / 256); };

  // degree normalization
  k_zero_f32 <<<blocks(N_NODES), B, 0, stream>>>(deg, N_NODES);
  k_deg_count<<<blocks(N_EDGES), B, 0, stream>>>(edst, deg);
  k_dinv     <<<blocks(N_NODES), B, 0, stream>>>(deg, dinv);
  k_edge_norm<<<blocks(N_EDGES), B, 0, stream>>>(esrc, edst, dinv, enorm);

  // GCN stack: real/padded dims (K pads -> mult 32, N pads -> mult 64)
  const int gKr[4] = {128, 100, 420, 140};
  const int gKp[4] = {128, 128, 448, 192};
  const int gNr[4] = {100, 420, 140, 140};
  const int gNp[4] = {128, 448, 192, 192};
  const float* asrc = x; int astride = 128, akreal = 128;
  for (int L = 0; L < 4; ++L) {
    int Kp = gKp[L], Np = gNp[L];
    long long atot = (long long)M_PAD * Kp;
    k_conv_a <<<blocks(atot), B, 0, stream>>>(asrc, abf, N_NODES, akreal, astride, Kp, (int)atot);
    k_conv_wt<<<blocks((long long)Np * Kp), B, 0, stream>>>(wg[L], wbf, gKr[L], gNr[L], Kp, Np);
    k_wmma_gemm<<<dim3(M_PAD / 128, Np / 64), B, 0, stream>>>(
        (const __bf16*)abf, (const __bf16*)wbf, h, Kp, Np);
    long long ntot = (long long)M_PAD * Np;
    k_zero_f32<<<blocks(ntot), B, 0, stream>>>(agg, (int)ntot);
    k_edge_agg<<<blocks((long long)N_EDGES * (Np >> 2)), B, 0, stream>>>(h, agg, esrc, edst, enorm, Np);
    k_combine <<<blocks((long long)N_NODES * Np), B, 0, stream>>>(h, agg, dinv, bg[L], xact, Np, gNr[L]);
    asrc = xact; astride = Np; akreal = Np;
  }

  // graph pooling (width padded to 192)
  k_zero_f32<<<blocks(N_GRAPHS * 192), B, 0, stream>>>(gA, N_GRAPHS * 192);
  k_pool    <<<blocks((long long)N_NODES * 192), B, 0, stream>>>(xact, batch, gA, 192);
  k_relu    <<<blocks(N_GRAPHS * 192), B, 0, stream>>>(gA, N_GRAPHS * 192);

  // dense stack (M = 1024)
  const int dKr[4] = {140, 260, 60, 180};
  const int dKp[4] = {192, 320, 64, 192};
  const int dNr[4] = {260, 60, 180, 100};
  const int dNp[4] = {320, 64, 192, 128};
  float* cur = gA; int cstride = 192;
  float* bufs[2] = {gB, gA};
  for (int L = 0; L < 4; ++L) {
    int Kp = dKp[L], Np = dNp[L];
    long long atot = (long long)N_GRAPHS * Kp;
    k_conv_a <<<blocks(atot), B, 0, stream>>>(cur, abf, N_GRAPHS, cstride, cstride, Kp, (int)atot);
    k_conv_wt<<<blocks((long long)Np * Kp), B, 0, stream>>>(wd[L], wbf, dKr[L], dNr[L], Kp, Np);
    k_wmma_gemm<<<dim3(N_GRAPHS / 128, Np / 64), B, 0, stream>>>(
        (const __bf16*)abf, (const __bf16*)wbf, h, Kp, Np);
    float* nxt = bufs[L & 1];
    k_bias_relu<<<blocks((long long)N_GRAPHS * Np), B, 0, stream>>>(h, bd[L], nxt, Np, dNr[L]);
    cur = nxt; cstride = Np;
  }

  // Antoine head: cur == gA, stride 128, 100 real features
  k_antoine<<<blocks(N_GRAPHS), B, 0, stream>>>(cur, want, bant, temp, mean, stdv, out);
}
